// SoftHausdorffLoss_69312182223151
// MI455X (gfx1250) — compile-verified, matrix operand order checked by reading, not run on hardware
//
#include <hip/hip_runtime.h>
#include <hip/hip_bf16.h>
#include <math.h>

#define Bb 2
#define Cc 3
#define Hh 112
#define Ww 112
#define Nn (Hh * Ww)       // 12544
#define Pp (Bb * Cc)       // 6
#define SORT_M 16384       // next pow2 >= Nn, for bitonic sort
#define INF_BITS 0x7F800000u
#define PAD_HI   0x00007800u   // f16(32768) in low half: pad |p|^2 > max real d2
#define ONES_H2  0x3C003C00u   // (f16)1.0 in both halves

typedef __attribute__((ext_vector_type(16))) _Float16 v16h;
typedef __attribute__((ext_vector_type(8)))  float    v8f;
typedef __attribute__((ext_vector_type(8)))  unsigned v8u;

__device__ __forceinline__ unsigned hd_pack2h(float lo, float hi) {
    _Float16 hl = (_Float16)lo, hh = (_Float16)hi;
    unsigned short ul = __builtin_bit_cast(unsigned short, hl);
    unsigned short uh = __builtin_bit_cast(unsigned short, hh);
    return (unsigned)ul | ((unsigned)uh << 16);
}

// ---------------------------------------------------------------- init
__global__ void hd_init_counters(int* seg_cnt, int* gt_cnt) {
    int t = threadIdx.x;
    if (t < Pp) { seg_cnt[t] = 0; gt_cnt[t] = 0; }
}

// ------------------------------------------------- argmax + compaction
// List entry (uint4): .x = packed f16 coords (px | py<<16)
//                     .y = packed f16 norm split (sqp_hi | sqp_lo<<16)
//                     .z = voxel index n (for the quantile gather)
__global__ void hd_argmax_compact(const float* __restrict__ x,
                                  const int*   __restrict__ y,
                                  int* seg_cnt, int* gt_cnt,
                                  uint4* seg_list, uint4* gt_list) {
    int idx = blockIdx.x * blockDim.x + threadIdx.x;
    if (idx >= Bb * Nn) return;
    int b = idx / Nn, n = idx - b * Nn;
    // argmax over channels, first-max tie-break (strict >)
    float best = x[(b * Cc + 0) * Nn + n];
    int cbest = 0;
#pragma unroll
    for (int c = 1; c < Cc; ++c) {
        float v = x[(b * Cc + c) * Nn + n];
        if (v > best) { best = v; cbest = c; }
    }

    int pi = n / Ww, pj = n % Ww;
    int sqp = pi * pi + pj * pj;
    uint4 e;
    e.x = hd_pack2h((float)pi, (float)pj);
    e.y = hd_pack2h((float)(sqp & ~15), (float)(sqp & 15));
    e.z = (unsigned)n;
    e.w = 0u;

    int ps = b * Cc + cbest;
    int ks = atomicAdd(&seg_cnt[ps], 1);
    seg_list[ps * Nn + ks] = e;

    int cg = y[b * Nn + n];
    int pg = b * Cc + cg;
    int kg = atomicAdd(&gt_cnt[pg], 1);
    gt_list[pg * Nn + kg] = e;
}

// ---------------------------------------------- WMMA min-distance core
// One wave per block. blockIdx = (query tile of 64 rows, direction, pair).
// d2(i,j) = |q|^2 + |p|^2 - 2 q.p from one v_wmma_f32_16x16x32_f16 with
// C = 0 by packing the norm terms into spare K slots:
//   K0: A=-2qx   B=px      K1: A=-2qy   B=py
//   K2: A=1      B=sqp_hi  K3: A=1      B=sqp_lo
//   K4: A=sqq_hi B=1       K5: A=sqq_lo B=1
// Coordinates <= 111 exact in f16; norms (<= 24642) split into a
// multiple-of-16 hi part (exact in f16 below 32768) and lo in [0,15];
// f32 accumulation of small integers -> d2 bit-exact and >= 0, so the
// running min runs on u32 bit patterns (v_min3_u32, no canonicalize).
// Two chunks per iteration; both B operands built before any WMMA, the
// 8 WMMAs grouped ahead of a sched_barrier(0), then 32 v_min3_u32 in
// t-major order so every consumer sits >= 4 co-executed instructions
// after its producing WMMA (no v_nop padding needed).
__global__ void __launch_bounds__(32)
hd_min_dist_wmma(const uint4* __restrict__ seg_list, const uint4* __restrict__ gt_list,
                 const int* __restrict__ seg_cnt,  const int* __restrict__ gt_cnt,
                 float* __restrict__ d_to_gt, float* __restrict__ d_to_seg) {
    const int lane = threadIdx.x;
    const int half = lane >> 4;          // 0: lanes 0-15, 1: lanes 16-31
    const int tile = blockIdx.x;         // 0..195, 64 query rows each
    const int dir  = blockIdx.y;         // 0: set = gt, 1: set = seg
    const int p    = blockIdx.z;         // pair index

    const uint4* list = (dir == 0) ? (gt_list + p * Nn) : (seg_list + p * Nn);
    const int    cnt  = (dir == 0) ? gt_cnt[p] : seg_cnt[p];
    float*       out  = ((dir == 0) ? d_to_gt : d_to_seg) + p * Nn;

    const int qbase = tile * 64;
    const bool data_lane = (half == 0);
    const int  sublane   = lane & 15;

    // Loop-invariant per-lane B constants.
    const unsigned pad1 = data_lane ? PAD_HI : 0u;   // invalid col: |p|^2 = 32768
    const unsigned ones = data_lane ? ONES_H2 : 0u;  // K4,K5 = 1,1

    // Per-subtile A operands. 16-bit A 16x32 layout: lanes 0-15 hold row
    // m = lane, elements 0..7 = K0..K7. Lanes 16-31 (K8..15) stay zero.
    v16h amat[4];
#pragma unroll
    for (int t = 0; t < 4; ++t) {
        v16h a;
#pragma unroll
        for (int e = 0; e < 16; ++e) a[e] = (_Float16)0.0f;
        if (data_lane) {
            int q = qbase + t * 16 + lane;
            int qi = q / Ww, qj = q % Ww;
            int sqq = qi * qi + qj * qj;
            a[0] = (_Float16)(float)(-2 * qi);
            a[1] = (_Float16)(float)(-2 * qj);
            a[2] = (_Float16)1.0f;
            a[3] = (_Float16)1.0f;
            a[4] = (_Float16)(float)(sqq & ~15);   // hi: multiple of 16
            a[5] = (_Float16)(float)(sqq & 15);    // lo: 0..15
        }
        amat[t] = a;
    }

    unsigned mins[4][8];
#pragma unroll
    for (int t = 0; t < 4; ++t)
#pragma unroll
        for (int r = 0; r < 8; ++r) mins[t][r] = INF_BITS;

    v8f czero;
#pragma unroll
    for (int r = 0; r < 8; ++r) czero[r] = 0.0f;

    const int nchunks = (cnt + 15) >> 4;

    // Prefetch first two chunks (clamped address: ws is always readable).
    int kA = sublane;
    int kB = 16 + sublane;
    bool vA = data_lane && (kA < cnt);
    bool vB = data_lane && (kB < cnt);
    uint4 eA = list[vA ? kA : 0];
    uint4 eB = list[vB ? kB : 0];

    int c = 0;
    for (; c + 1 < nchunks; c += 2) {
        // Prefetch chunks c+2 / c+3 (latency hidden under this iteration).
        int kA2 = (c + 2) * 16 + sublane;
        int kB2 = (c + 3) * 16 + sublane;
        bool vA2 = data_lane && (kA2 < cnt);
        bool vB2 = data_lane && (kB2 < cnt);
        uint4 eA2 = list[vA2 ? kA2 : 0];
        uint4 eB2 = list[vB2 ? kB2 : 0];

        // Build both B operands before any WMMA (distinct live registers).
        v8u bwA = {};
        bwA[0] = vA ? eA.x : 0u;
        bwA[1] = vA ? eA.y : pad1;
        bwA[2] = ones;
        v16h bA = __builtin_bit_cast(v16h, bwA);

        v8u bwB = {};
        bwB[0] = vB ? eB.x : 0u;
        bwB[1] = vB ? eB.y : pad1;
        bwB[2] = ones;
        v16h bB = __builtin_bit_cast(v16h, bwB);

        // 8 independent-destination WMMAs, grouped ahead of the fence.
        v8f dA0 = __builtin_amdgcn_wmma_f32_16x16x32_f16(
            false, amat[0], false, bA, (short)0, czero, false, false);
        v8f dB0 = __builtin_amdgcn_wmma_f32_16x16x32_f16(
            false, amat[0], false, bB, (short)0, czero, false, false);
        v8f dA1 = __builtin_amdgcn_wmma_f32_16x16x32_f16(
            false, amat[1], false, bA, (short)0, czero, false, false);
        v8f dB1 = __builtin_amdgcn_wmma_f32_16x16x32_f16(
            false, amat[1], false, bB, (short)0, czero, false, false);
        v8f dA2 = __builtin_amdgcn_wmma_f32_16x16x32_f16(
            false, amat[2], false, bA, (short)0, czero, false, false);
        v8f dB2 = __builtin_amdgcn_wmma_f32_16x16x32_f16(
            false, amat[2], false, bB, (short)0, czero, false, false);
        v8f dA3 = __builtin_amdgcn_wmma_f32_16x16x32_f16(
            false, amat[3], false, bA, (short)0, czero, false, false);
        v8f dB3 = __builtin_amdgcn_wmma_f32_16x16x32_f16(
            false, amat[3], false, bB, (short)0, czero, false, false);

        // Fence: no min3 may move up between the WMMAs (and no WMMA may
        // sink into the min block) -> all consumers outside the hazard
        // window, no v_nop padding.
        __builtin_amdgcn_sched_barrier(0);

        // v_min3_u32: min(dA, dB, running), t-major so the t=3 group
        // (consumers of the last WMMAs) runs after 24 other min3 ops.
#pragma unroll
        for (int r = 0; r < 8; ++r) {
            unsigned a0 = __float_as_uint(dA0[r]), b0 = __float_as_uint(dB0[r]);
            unsigned m0 = (a0 < b0) ? a0 : b0;
            mins[0][r] = (m0 < mins[0][r]) ? m0 : mins[0][r];
        }
#pragma unroll
        for (int r = 0; r < 8; ++r) {
            unsigned a1 = __float_as_uint(dA1[r]), b1 = __float_as_uint(dB1[r]);
            unsigned m1 = (a1 < b1) ? a1 : b1;
            mins[1][r] = (m1 < mins[1][r]) ? m1 : mins[1][r];
        }
#pragma unroll
        for (int r = 0; r < 8; ++r) {
            unsigned a2 = __float_as_uint(dA2[r]), b2 = __float_as_uint(dB2[r]);
            unsigned m2 = (a2 < b2) ? a2 : b2;
            mins[2][r] = (m2 < mins[2][r]) ? m2 : mins[2][r];
        }
#pragma unroll
        for (int r = 0; r < 8; ++r) {
            unsigned a3 = __float_as_uint(dA3[r]), b3 = __float_as_uint(dB3[r]);
            unsigned m3 = (a3 < b3) ? a3 : b3;
            mins[3][r] = (m3 < mins[3][r]) ? m3 : mins[3][r];
        }

        eA = eA2; vA = vA2;
        eB = eB2; vB = vB2;
    }

    if (c < nchunks) {   // odd tail chunk
        v8u bwA = {};
        bwA[0] = vA ? eA.x : 0u;
        bwA[1] = vA ? eA.y : pad1;
        bwA[2] = ones;
        v16h bA = __builtin_bit_cast(v16h, bwA);

        v8f dA0 = __builtin_amdgcn_wmma_f32_16x16x32_f16(
            false, amat[0], false, bA, (short)0, czero, false, false);
        v8f dA1 = __builtin_amdgcn_wmma_f32_16x16x32_f16(
            false, amat[1], false, bA, (short)0, czero, false, false);
        v8f dA2 = __builtin_amdgcn_wmma_f32_16x16x32_f16(
            false, amat[2], false, bA, (short)0, czero, false, false);
        v8f dA3 = __builtin_amdgcn_wmma_f32_16x16x32_f16(
            false, amat[3], false, bA, (short)0, czero, false, false);
#pragma unroll
        for (int r = 0; r < 8; ++r) {
            unsigned u0 = __float_as_uint(dA0[r]);
            unsigned u1 = __float_as_uint(dA1[r]);
            unsigned u2 = __float_as_uint(dA2[r]);
            unsigned u3 = __float_as_uint(dA3[r]);
            mins[0][r] = (u0 < mins[0][r]) ? u0 : mins[0][r];
            mins[1][r] = (u1 < mins[1][r]) ? u1 : mins[1][r];
            mins[2][r] = (u2 < mins[2][r]) ? u2 : mins[2][r];
            mins[3][r] = (u3 < mins[3][r]) ? u3 : mins[3][r];
        }
    }

    // Column-min across the 16 lanes of each half, then sqrt + store.
#pragma unroll
    for (int t = 0; t < 4; ++t) {
#pragma unroll
        for (int r = 0; r < 8; ++r) {
            unsigned m = mins[t][r];
#pragma unroll
            for (int off = 1; off < 16; off <<= 1) {
                unsigned o = __shfl_xor(m, off, 32);
                m = (o < m) ? o : m;
            }
            if (sublane == 0) {
                int q = qbase + t * 16 + r + 8 * half;
                out[q] = sqrtf(__uint_as_float(m));
            }
        }
    }
}

// ----------------------------------------- masked quantile via bitonic
__global__ void hd_quantile(const uint4* __restrict__ seg_list, const uint4* __restrict__ gt_list,
                            const int* __restrict__ seg_cnt,  const int* __restrict__ gt_cnt,
                            const float* __restrict__ d_to_gt, const float* __restrict__ d_to_seg,
                            float* __restrict__ quant) {
    extern __shared__ float s[];
    const int bx = blockIdx.x;            // 0..11
    const int p = bx >> 1, which = bx & 1;
    // which==0: hd1 = quantile of d_to_gt over seg points
    // which==1: hd2 = quantile of d_to_seg over gt points
    const int cnt = which ? gt_cnt[p] : seg_cnt[p];
    const uint4* list = (which ? gt_list : seg_list) + p * Nn;
    const float* dv = (which ? d_to_seg : d_to_gt) + p * Nn;
    const int tid = threadIdx.x, nt = blockDim.x;

    for (int i = tid; i < SORT_M; i += nt)
        s[i] = (i < cnt) ? dv[list[i].z] : 3.4e38f;
    __syncthreads();

    for (int k = 2; k <= SORT_M; k <<= 1) {
        for (int j = k >> 1; j > 0; j >>= 1) {
            for (int i = tid; i < SORT_M; i += nt) {
                int ixj = i ^ j;
                if (ixj > i) {
                    float a = s[i], b = s[ixj];
                    bool up = ((i & k) == 0);
                    if (up ? (a > b) : (a < b)) { s[i] = b; s[ixj] = a; }
                }
            }
            __syncthreads();
        }
    }

    if (tid == 0) {
        float v = 0.0f;
        if (cnt > 0) {
            float pos = 0.95f * (float)(cnt - 1);
            int lo = (int)floorf(pos);
            int hi = (int)ceilf(pos);
            float frac = pos - (float)lo;
            v = s[lo] * (1.0f - frac) + s[hi] * frac;
        }
        quant[bx] = v;
    }
}

// ----------------------------------------------------------- finalize
__global__ void hd_finalize(const int* seg_cnt, const int* gt_cnt,
                            const float* quant, float* out) {
    if (threadIdx.x == 0 && blockIdx.x == 0) {
        float acc = 0.0f;
        for (int p = 0; p < Pp; ++p) {
            if (seg_cnt[p] > 0 && gt_cnt[p] > 0)
                acc += fmaxf(quant[2 * p], quant[2 * p + 1]);
        }
        out[0] = acc / (float)Pp;
    }
}

// ------------------------------------------------------------- launch
extern "C" void kernel_launch(void* const* d_in, const int* in_sizes, int n_in,
                              void* d_out, int out_size, void* d_ws, size_t ws_size,
                              hipStream_t stream) {
    const float* x = (const float*)d_in[0];   // [B,C,H,W] f32
    const int*   y = (const int*)d_in[1];     // [B,H,W]   i32
    float* out = (float*)d_out;

    char* ws = (char*)d_ws;
    size_t off = 0;
    int* seg_cnt = (int*)(ws + off);   off += 256;
    int* gt_cnt  = (int*)(ws + off);   off += 256;
    float* quant = (float*)(ws + off); off += 256;
    uint4* seg_list = (uint4*)(ws + off); off += (size_t)Pp * Nn * sizeof(uint4);
    uint4* gt_list  = (uint4*)(ws + off); off += (size_t)Pp * Nn * sizeof(uint4);
    float* d_to_gt  = (float*)(ws + off); off += (size_t)Pp * Nn * sizeof(float);
    float* d_to_seg = (float*)(ws + off); off += (size_t)Pp * Nn * sizeof(float);

    hd_init_counters<<<1, 32, 0, stream>>>(seg_cnt, gt_cnt);

    hd_argmax_compact<<<(Bb * Nn + 255) / 256, 256, 0, stream>>>(
        x, y, seg_cnt, gt_cnt, seg_list, gt_list);

    // 196 tiles of 64 query rows x 2 directions x 6 pairs, one wave each
    hd_min_dist_wmma<<<dim3(Nn / 64, 2, Pp), 32, 0, stream>>>(
        seg_list, gt_list, seg_cnt, gt_cnt, d_to_gt, d_to_seg);

    hd_quantile<<<12, 1024, SORT_M * sizeof(float), stream>>>(
        seg_list, gt_list, seg_cnt, gt_cnt, d_to_gt, d_to_seg, quant);

    hd_finalize<<<1, 1, 0, stream>>>(seg_cnt, gt_cnt, quant, out);
}
